// Transformer_26302379720772
// MI455X (gfx1250) — matrix operand order repaired and checked by reading, not compiled, not run on hardware
//
#include <hip/hip_runtime.h>
#include <hip/hip_bf16.h>
#include <math.h>

typedef __attribute__((ext_vector_type(16))) __bf16 v16bf;
typedef __attribute__((ext_vector_type(8)))  float  v8f;
typedef __attribute__((ext_vector_type(8)))  int    v8i;
typedef __attribute__((ext_vector_type(4)))  int    v4i;
typedef __attribute__((ext_vector_type(4)))  unsigned int u32x4;
typedef __attribute__((ext_vector_type(8)))  unsigned int u32x8;
typedef unsigned short u16;

// ---------------- bf16 helpers ----------------
__device__ __forceinline__ u16 f2bf(float f) {
    unsigned int u = __float_as_uint(f);
    u += 0x7FFFu + ((u >> 16) & 1u);   // round-to-nearest-even
    return (u16)(u >> 16);
}

// ---------------- WMMA fragment loaders ----------------
// A fragment (16x32 bf16, row-major source [M,K]):
// lane m = lane&15 is row; element e -> K = 16*hi + e  (16 contiguous bf16)
__device__ __forceinline__ v16bf load_afrag(const u16* A, int lda, int row0, int k0, int lane) {
    int m = lane & 15, hi = lane >> 4;
    const u16* p = A + (size_t)(row0 + m) * lda + k0 + 16 * hi;
    v8i r = *(const v8i*)p;
    return __builtin_bit_cast(v16bf, r);
}

// B fragment (32x16 bf16) from transposed storage Bt[N,K] row-major:
// lane n = lane&15 is column; element e -> K = 16*hi + e  (16 contiguous bf16)
__device__ __forceinline__ v16bf load_bfrag(const u16* Bt, int ldb, int col0, int k0, int lane) {
    int n = lane & 15, hi = lane >> 4;
    const u16* p = Bt + (size_t)(col0 + n) * ldb + k0 + 16 * hi;
    v8i r = *(const v8i*)p;
    return __builtin_bit_cast(v16bf, r);
}

__device__ __forceinline__ v8f wmma_bf(v16bf a, v16bf b, v8f c) {
    return __builtin_amdgcn_wmma_f32_16x16x32_bf16(false, a, false, b, (short)0, c, false, false);
}

// ---------------- Tensor Data Mover: stage a 64-row x 384-col bf16 tile into LDS ----------------
// D# per CDNA5 ISA ch8: group0 {count, lds_addr, global_addr(57b), type=2},
// group1 {data_size=2B, tensor_dim0=384, tensor_dim1=64, tile_dim0=384, tile_dim1=64,
//         tensor_dim0_stride=ldb}.  2D tile -> groups 2/3 NULL (2-operand form).
__device__ __forceinline__ void tdm_load_tile(unsigned lds_off, const u16* gptr, int ldbElems) {
    unsigned long long ga = (unsigned long long)(size_t)gptr;
    u32x4 g0;
    g0[0] = 1u;                                            // count=1, user mode, no gather
    g0[1] = lds_off;                                       // LDS byte address
    g0[2] = (unsigned)(ga & 0xFFFFFFFFu);                  // global_addr[31:0]
    g0[3] = (unsigned)((ga >> 32) & 0x1FFFFFFu) | (2u << 30);  // addr[56:32] | type=2
    const unsigned td0 = 384u, td1 = 64u, tile0 = 384u, tile1 = 64u;
    unsigned long long s0 = (unsigned long long)(unsigned)ldbElems;  // dim0 stride (elems)
    u32x8 g1;
    g1[0] = (1u << 16);                                    // data_size=1 (2 bytes)
    g1[1] = (td0 & 0xFFFFu) << 16;                         // tensor_dim0[15:0] @ bit48
    g1[2] = ((td0 >> 16) & 0xFFFFu) | ((td1 & 0xFFFFu) << 16);
    g1[3] = ((td1 >> 16) & 0xFFFFu) | ((tile0 & 0xFFFFu) << 16);
    g1[4] = (tile1 & 0xFFFFu);                             // tile_dim2 = 0
    g1[5] = (unsigned)(s0 & 0xFFFFFFFFu);
    g1[6] = (unsigned)((s0 >> 32) & 0xFFFFu);              // tensor_dim1_stride = 0
    g1[7] = 0u;
    asm volatile("tensor_load_to_lds %0, %1" :: "s"(g0), "s"(g1) : "memory");
}

// One wave computes a 16(M) x 64(N) strip; the block's B tile is staged in LDS by the TDM.
// C layout: elem r of v8f -> row = r + 8*hi, col = lane&15.
template <int KTOT>
__device__ __forceinline__ void gemm_core_lds(const u16* A, int lda, const u16* BtG, int ldb,
                                              int m0, int n0, int lane, int wv,
                                              u16* bsh, unsigned bsh_off, v8f acc[4]) {
    for (int kc = 0; kc < KTOT; kc += 384) {
        __syncthreads();                       // all waves done with previous chunk
        if (wv == 0) {
            tdm_load_tile(bsh_off, BtG + (size_t)n0 * ldb + kc, ldb);
            __builtin_amdgcn_s_wait_tensorcnt(0);
        }
        __syncthreads();                       // tile visible to all waves
        for (int k0 = 0; k0 < 384; k0 += 32) {
            v16bf a = load_afrag(A, lda, m0, kc + k0, lane);
#pragma unroll
            for (int j = 0; j < 4; ++j) {
                v16bf b = load_bfrag(bsh, 384, 16 * j, k0, lane);   // block-local cols
                acc[j] = wmma_bf(a, b, acc[j]);
            }
        }
    }
}

// ---------------- block reductions (384 threads = 12 waves of 32) ----------------
__device__ __forceinline__ void block_sum2(float a, float b, float* red, int tid,
                                           float& oa, float& ob) {
    __syncthreads();
#pragma unroll
    for (int o = 16; o > 0; o >>= 1) { a += __shfl_xor(a, o, 32); b += __shfl_xor(b, o, 32); }
    int w = tid >> 5;
    if ((tid & 31) == 0) { red[w] = a; red[w + 16] = b; }
    __syncthreads();
    if (tid < 32) {
        float x = (tid < 12) ? red[tid] : 0.f;
        float y = (tid < 12) ? red[16 + tid] : 0.f;
#pragma unroll
        for (int o = 16; o > 0; o >>= 1) { x += __shfl_xor(x, o, 32); y += __shfl_xor(y, o, 32); }
        if (tid == 0) { red[0] = x; red[16] = y; }
    }
    __syncthreads();
    oa = red[0]; ob = red[16];
}

// ---------------- kernels ----------------
// x[B,C,N] -> t[tok,384] fp32 and a[tok,384] bf16 after double layernorm
__global__ void k_prep(const float* __restrict__ x,
                       const float* __restrict__ g1, const float* __restrict__ b1,
                       const float* __restrict__ ga, const float* __restrict__ ba,
                       float* __restrict__ t, u16* __restrict__ a) {
    __shared__ float red[32];
    int tok = blockIdx.x, c = threadIdx.x;        // 8192 blocks x 384 threads
    int b = tok >> 10, n = tok & 1023;
    float v = x[((size_t)b * 384 + c) * 1024 + n];
    t[(size_t)tok * 384 + c] = v;
    float s, ss;
    block_sum2(v, v * v, red, c, s, ss);
    float mu = s * (1.f / 384.f), var = ss * (1.f / 384.f) - mu * mu;
    float y = (v - mu) * rsqrtf(var + 1e-5f) * g1[c] + b1[c];
    block_sum2(y, y * y, red, c, s, ss);
    mu = s * (1.f / 384.f); var = ss * (1.f / 384.f) - mu * mu;
    float z = (y - mu) * rsqrtf(var + 1e-5f) * ga[c] + ba[c];
    a[(size_t)tok * 384 + c] = f2bf(z);
}

// plain layernorm: fp32 [tok,384] -> bf16 [tok,384]
__global__ void k_ln(const float* __restrict__ in, const float* __restrict__ g,
                     const float* __restrict__ bb, u16* __restrict__ out) {
    __shared__ float red[32];
    int tok = blockIdx.x, c = threadIdx.x;
    float v = in[(size_t)tok * 384 + c];
    float s, ss;
    block_sum2(v, v * v, red, c, s, ss);
    float mu = s * (1.f / 384.f), var = ss * (1.f / 384.f) - mu * mu;
    out[(size_t)tok * 384 + c] = f2bf((v - mu) * rsqrtf(var + 1e-5f) * g[c] + bb[c]);
}

// transpose + convert weight [rows(in), cols(out)] fp32 -> wt[out, in] bf16
__global__ void k_wt(const float* __restrict__ w, u16* __restrict__ wt, int rows, int cols) {
    int idx = blockIdx.x * 256 + threadIdx.x;
    if (idx < rows * cols) {
        int r = idx / cols, c = idx - r * cols;
        wt[(size_t)c * rows + r] = f2bf(w[idx]);
    }
}

// qkv GEMM: a[8192,384] x wqkvT[1152,384] ; scatter to padded q/k [B,H,N,64] and vT [B,H,64,N]
__global__ __launch_bounds__(256) void k_gemm_qkv(const u16* __restrict__ A, const u16* __restrict__ Bt,
                                                  u16* __restrict__ qp, u16* __restrict__ kp,
                                                  u16* __restrict__ vtp) {
    __shared__ __align__(16) u16 bsh[64 * 384];
    int lane = threadIdx.x & 31, wv = threadIdx.x >> 5;
    int m0 = blockIdx.x * 128 + wv * 16;
    int n0 = blockIdx.y * 64;
    v8f acc[4];
#pragma unroll
    for (int j = 0; j < 4; ++j)
#pragma unroll
        for (int r = 0; r < 8; ++r) acc[j][r] = 0.f;
    gemm_core_lds<384>(A, 384, Bt, 384, m0, n0, lane, wv, bsh, (unsigned)(size_t)(void*)bsh, acc);
    int hi = lane >> 4, lc = lane & 15;
#pragma unroll
    for (int j = 0; j < 4; ++j)
#pragma unroll
        for (int r = 0; r < 8; ++r) {
            int row = m0 + r + 8 * hi;
            int col = n0 + 16 * j + lc;
            int b = row >> 10, n = row & 1023;
            int which = col / 384, wi = col - which * 384;
            int h = wi / 48, dc = wi - h * 48;
            u16 val = f2bf(acc[j][r]);
            if (which == 0)
                qp[((size_t)(b * 8 + h) * 1024 + n) * 64 + dc] = val;
            else if (which == 1)
                kp[((size_t)(b * 8 + h) * 1024 + n) * 64 + dc] = val;
            else
                vtp[((size_t)(b * 8 + h) * 64 + dc) * 1024 + n] = val;
        }
}

// flash attention: one wave = 16 queries of one (b,h); loops over 1024 keys in 32-key tiles
__global__ __launch_bounds__(256) void k_attn(const u16* __restrict__ qp, const u16* __restrict__ kp,
                                              const u16* __restrict__ vtp, u16* __restrict__ o) {
    __shared__ __align__(16) u16 pst[8][512];   // per-wave 16x32 bf16 prob staging
    int lane = threadIdx.x & 31, wv = threadIdx.x >> 5;
    int bh = blockIdx.x;                 // b*8 + h
    int b = bh >> 3, h = bh & 7;
    int m0 = blockIdx.y * 128 + wv * 16; // query tile start
    const u16* q  = qp  + (size_t)bh * 1024 * 64;
    const u16* kk = kp  + (size_t)bh * 1024 * 64;
    const u16* vt = vtp + (size_t)bh * 64 * 1024;
    u16* ps = pst[wv];
    const float scale = 0.14433756729740643f;   // 48^-0.5
    v16bf qa0 = load_afrag(q, 64, m0, 0, lane);
    v16bf qa1 = load_afrag(q, 64, m0, 32, lane);
    v8f acc[3];
    float mrow[8], lrow[8];
#pragma unroll
    for (int r = 0; r < 8; ++r) {
        acc[0][r] = 0.f; acc[1][r] = 0.f; acc[2][r] = 0.f;
        mrow[r] = -3.0e38f; lrow[r] = 0.f;
    }
    int hi = lane >> 4, lc = lane & 15;
    for (int kt = 0; kt < 1024; kt += 32) {
        v8f s0, s1;
#pragma unroll
        for (int r = 0; r < 8; ++r) { s0[r] = 0.f; s1[r] = 0.f; }
        s0 = wmma_bf(qa0, load_bfrag(kk, 64, kt,      0,  lane), s0);
        s0 = wmma_bf(qa1, load_bfrag(kk, 64, kt,      32, lane), s0);
        s1 = wmma_bf(qa0, load_bfrag(kk, 64, kt + 16, 0,  lane), s1);
        s1 = wmma_bf(qa1, load_bfrag(kk, 64, kt + 16, 32, lane), s1);
#pragma unroll
        for (int r = 0; r < 8; ++r) {
            float a = s0[r] * scale, c = s1[r] * scale;
            float mx = fmaxf(a, c);
            mx = fmaxf(mx, __shfl_xor(mx, 1, 32));
            mx = fmaxf(mx, __shfl_xor(mx, 2, 32));
            mx = fmaxf(mx, __shfl_xor(mx, 4, 32));
            mx = fmaxf(mx, __shfl_xor(mx, 8, 32));
            float nm = fmaxf(mrow[r], mx);
            float f  = __expf(mrow[r] - nm);
            float p0 = __expf(a - nm), p1 = __expf(c - nm);
            float sm = p0 + p1;
            sm += __shfl_xor(sm, 1, 32);
            sm += __shfl_xor(sm, 2, 32);
            sm += __shfl_xor(sm, 4, 32);
            sm += __shfl_xor(sm, 8, 32);
            lrow[r] = lrow[r] * f + sm;
            mrow[r] = nm;
            acc[0][r] *= f; acc[1][r] *= f; acc[2][r] *= f;
            int rr = r + 8 * hi;                 // row owned by this element
            ps[rr * 32 + lc]      = f2bf(p0);    // keys kt .. kt+15
            ps[rr * 32 + 16 + lc] = f2bf(p1);    // keys kt+16 .. kt+31
        }
        asm volatile("s_wait_dscnt 0" ::: "memory");
        v16bf pa = load_afrag(ps, 32, 0, 0, lane);   // re-layout C->A via LDS
#pragma unroll
        for (int j = 0; j < 3; ++j)                  // only v-cols 0..47 (48..63 are zero pad)
            acc[j] = wmma_bf(pa, load_bfrag(vt, 1024, 16 * j, kt, lane), acc[j]);
    }
#pragma unroll
    for (int r = 0; r < 8; ++r) {
        float inv = 1.0f / lrow[r];
        int tok = b * 1024 + m0 + r + 8 * hi;
#pragma unroll
        for (int j = 0; j < 3; ++j)
            o[(size_t)tok * 384 + h * 48 + 16 * j + lc] = f2bf(acc[j][r] * inv);
    }
}

// out projection + residual in place: t[.,.] += o @ woutT
__global__ __launch_bounds__(256) void k_gemm_attnout(const u16* __restrict__ A, const u16* __restrict__ Bt,
                                                      float* __restrict__ t) {
    __shared__ __align__(16) u16 bsh[64 * 384];
    int lane = threadIdx.x & 31, wv = threadIdx.x >> 5;
    int m0 = blockIdx.x * 128 + wv * 16;
    int n0 = blockIdx.y * 64;
    v8f acc[4];
#pragma unroll
    for (int j = 0; j < 4; ++j)
#pragma unroll
        for (int r = 0; r < 8; ++r) acc[j][r] = 0.f;
    gemm_core_lds<384>(A, 384, Bt, 384, m0, n0, lane, wv, bsh, (unsigned)(size_t)(void*)bsh, acc);
    int hi = lane >> 4, lc = lane & 15;
#pragma unroll
    for (int j = 0; j < 4; ++j)
#pragma unroll
        for (int r = 0; r < 8; ++r) {
            size_t idx = (size_t)(m0 + r + 8 * hi) * 384 + n0 + 16 * j + lc;
            t[idx] = t[idx] + acc[j][r];
        }
}

// MLP fc1 + exact GELU -> bf16 hidden (chunk of 2048 rows)
__global__ __launch_bounds__(256) void k_gemm_mlp1(const u16* __restrict__ A, const u16* __restrict__ Bt,
                                                   const float* __restrict__ b1, u16* __restrict__ hbuf) {
    __shared__ __align__(16) u16 bsh[64 * 384];
    int lane = threadIdx.x & 31, wv = threadIdx.x >> 5;
    int m0 = blockIdx.x * 128 + wv * 16;
    int n0 = blockIdx.y * 64;
    v8f acc[4];
#pragma unroll
    for (int j = 0; j < 4; ++j)
#pragma unroll
        for (int r = 0; r < 8; ++r) acc[j][r] = 0.f;
    gemm_core_lds<384>(A, 384, Bt, 384, m0, n0, lane, wv, bsh, (unsigned)(size_t)(void*)bsh, acc);
    int hi = lane >> 4, lc = lane & 15;
#pragma unroll
    for (int j = 0; j < 4; ++j)
#pragma unroll
        for (int r = 0; r < 8; ++r) {
            int row = m0 + r + 8 * hi;
            int col = n0 + 16 * j + lc;
            float xv = acc[j][r] + b1[col];
            float ge = 0.5f * xv * (1.0f + erff(xv * 0.70710678118654752f));
            hbuf[(size_t)row * 1536 + col] = f2bf(ge);
        }
}

// MLP fc2 + bias + residual; write final output transposed back to [B, C, H*W]
__global__ __launch_bounds__(256) void k_gemm_mlp2(const u16* __restrict__ A, const u16* __restrict__ Bt,
                                                   const float* __restrict__ b2, const float* __restrict__ t2,
                                                   float* __restrict__ dout, int rowBase) {
    __shared__ __align__(16) u16 bsh[64 * 384];
    int lane = threadIdx.x & 31, wv = threadIdx.x >> 5;
    int m0 = blockIdx.x * 128 + wv * 16;
    int n0 = blockIdx.y * 64;
    v8f acc[4];
#pragma unroll
    for (int j = 0; j < 4; ++j)
#pragma unroll
        for (int r = 0; r < 8; ++r) acc[j][r] = 0.f;
    gemm_core_lds<1536>(A, 1536, Bt, 1536, m0, n0, lane, wv, bsh, (unsigned)(size_t)(void*)bsh, acc);
    int hi = lane >> 4, lc = lane & 15;
#pragma unroll
    for (int j = 0; j < 4; ++j)
#pragma unroll
        for (int r = 0; r < 8; ++r) {
            int row = m0 + r + 8 * hi;               // chunk-local row
            int col = n0 + 16 * j + lc;
            float val = acc[j][r] + b2[col] + t2[(size_t)row * 384 + col];
            int rg = rowBase + row;
            int b = rg >> 10, n = rg & 1023;
            dout[((size_t)b * 384 + col) * 1024 + n] = val;
        }
}

// ---------------- host launcher ----------------
extern "C" void kernel_launch(void* const* d_in, const int* in_sizes, int n_in,
                              void* d_out, int out_size, void* d_ws, size_t ws_size,
                              hipStream_t stream) {
    (void)in_sizes; (void)n_in; (void)out_size; (void)ws_size;
    const float* x    = (const float*)d_in[0];
    const float* ln1g = (const float*)d_in[1];
    const float* ln1b = (const float*)d_in[2];
    const float* lnag = (const float*)d_in[3];
    const float* lnab = (const float*)d_in[4];
    const float* wqkv = (const float*)d_in[5];
    const float* wout = (const float*)d_in[6];
    const float* ln2g = (const float*)d_in[7];
    const float* ln2b = (const float*)d_in[8];
    const float* w1   = (const float*)d_in[9];
    const float* b1   = (const float*)d_in[10];
    const float* w2   = (const float*)d_in[11];
    const float* b2   = (const float*)d_in[12];
    float* out = (float*)d_out;

    char* ws = (char*)d_ws;
    float* t     = (float*)(ws + 0);          // 8192*384*4  = 12,582,912
    u16*   a     = (u16*)(ws + 12582912);     // 8192*384*2  =  6,291,456
    u16*   wqkvT = (u16*)(ws + 18874368);     // 1152*384*2  =    884,736
    u16*   woutT = (u16*)(ws + 19759104);     // 384*384*2   =    294,912
    u16*   w1T   = (u16*)(ws + 20054016);     // 1536*384*2  =  1,179,648
    u16*   w2T   = (u16*)(ws + 21233664);     // 384*1536*2  =  1,179,648
    u16*   qp    = (u16*)(ws + 22413312);     // 8*8*1024*64*2 = 8,388,608
    u16*   kp    = (u16*)(ws + 30801920);     // 8,388,608
    u16*   vtp   = (u16*)(ws + 39190528);     // 8,388,608
    u16*   o     = (u16*)(ws + 47579136);     // 6,291,456
    u16*   m     = (u16*)(ws + 53870592);     // 6,291,456
    u16*   hbuf  = (u16*)(ws + 60162048);     // 2048*1536*2 = 6,291,456  (end ~66.5 MB)

    // zero q/k/v padded buffers (pad lanes 48..63 must be zero each call)
    hipMemsetAsync(qp, 0, (size_t)3 * 8388608, stream);

    // weight convert+transpose to bf16 [out,in]
    k_wt<<<dim3((384 * 1152 + 255) / 256), 256, 0, stream>>>(wqkv, wqkvT, 384, 1152);
    k_wt<<<dim3((384 * 384 + 255) / 256), 256, 0, stream>>>(wout, woutT, 384, 384);
    k_wt<<<dim3((384 * 1536 + 255) / 256), 256, 0, stream>>>(w1, w1T, 384, 1536);
    k_wt<<<dim3((1536 * 384 + 255) / 256), 256, 0, stream>>>(w2, w2T, 1536, 384);

    // tokens + double pre-norm
    k_prep<<<8192, 384, 0, stream>>>(x, ln1g, ln1b, lnag, lnab, t, a);

    // qkv projection (scatter into padded head layout)
    k_gemm_qkv<<<dim3(64, 18), 256, 0, stream>>>(a, wqkvT, qp, kp, vtp);

    // flash attention
    k_attn<<<dim3(64, 8), 256, 0, stream>>>(qp, kp, vtp, o);

    // out projection + residual (t becomes t2 in place)
    k_gemm_attnout<<<dim3(64, 6), 256, 0, stream>>>(o, woutT, t);

    // LN2
    k_ln<<<8192, 384, 0, stream>>>(t, ln2g, ln2b, m);

    // MLP, chunked over rows to bound workspace
    for (int cb = 0; cb < 4; ++cb) {
        int rowBase = cb * 2048;
        k_gemm_mlp1<<<dim3(16, 24), 256, 0, stream>>>(m + (size_t)rowBase * 384, w1T, b1, hbuf);
        k_gemm_mlp2<<<dim3(16, 6), 256, 0, stream>>>(hbuf, w2T, b2, t + (size_t)rowBase * 384, out, rowBase);
    }
}
